// OmniAnomaly_Encoder_54735063220645
// MI455X (gfx1250) — matrix operand order repaired and checked by reading, not compiled, not run on hardware
//
#include <hip/hip_runtime.h>
#include <hip/hip_bf16.h>
#include <math.h>

typedef __attribute__((ext_vector_type(16))) __bf16 v16bf;
typedef __attribute__((ext_vector_type(8)))  float  v8f;

#define B_  256
#define S_  256
#define F_  64
#define H_  512
#define L_  16
#define KF_ 20
#define G3H 1536
#define NTOT (B_ * S_ * L_)   // 1048576 elements per output tensor

// ---- packed bf16 weight-fragment offsets inside d_ws (ushort elements) ----
// layout per matrix: [mat][ntile][ktile][lane(32)][kr(16)] ; each lane's 16
// bf16 = 32 contiguous bytes = two global_load_b128 per WMMA B fragment.
#define OFF_W   0         //  96 ntiles x  2 ktiles
#define OFF_U   98304     //  96 ntiles x 16 ktiles
#define OFF_WD  884736    //  32 ntiles x 16 ktiles
#define OFF_WM  1146880   //   1 ntile  x 16 ktiles
#define OFF_WV  1155072
#define OFF_F1  1163264   //  20 mats x 1 x 1 (K padded 16->32 with zeros)
#define OFF_F2  1173504

static __device__ __forceinline__ unsigned short f2bf(float f) {
  unsigned int x = __builtin_bit_cast(unsigned int, f);
  x += 0x7fffu + ((x >> 16) & 1u);          // round-to-nearest-even
  return (unsigned short)(x >> 16);
}
static __device__ __forceinline__ float bf2f(unsigned short h) {
  unsigned int x = ((unsigned int)h) << 16;
  return __builtin_bit_cast(float, x);
}

struct Frag {
  union { v16bf v; uint4 q[2]; unsigned short us[16]; };
};

// B fragment: 32 contiguous bytes per lane from pre-swizzled scratch.
static __device__ __forceinline__ void loadB_g(Frag& f, const unsigned short* p) {
  f.q[0] = *(const uint4*)(p);
  f.q[1] = *(const uint4*)(p + 8);
}
// A fragment from row-major bf16 LDS: caller passes &lds[m*stride + kt*32 + khalf*8];
// VGPR0..3 hold K = kbase..kbase+7, VGPR4..7 hold K = kbase+16..kbase+23.
static __device__ __forceinline__ void loadA_lds(Frag& f, const unsigned short* p) {
  f.q[0] = *(const uint4*)(p);
  f.q[1] = *(const uint4*)(p + 16);
}
static __device__ __forceinline__ v8f wmma_bf16(const Frag& a, const Frag& b, v8f c) {
  return __builtin_amdgcn_wmma_f32_16x16x32_bf16(false, a.v, false, b.v,
                                                 (short)0, c, false, false);
}
static __device__ __forceinline__ float sigmoidf_(float x) {
  return 1.0f / (1.0f + __expf(-x));
}

// ---------------- weight packing: fp32 row-major [K x N] -> bf16 fragments ----
__global__ __launch_bounds__(256) void pack_frag_kernel(
    const float* __restrict__ src, unsigned short* __restrict__ dst,
    int Kdim, int Ndim, int ktiles, int ntiles, int nmats, long srcMatStride)
{
  long total = (long)nmats * ntiles * ktiles * 512;
  for (long idx = (long)blockIdx.x * blockDim.x + threadIdx.x; idx < total;
       idx += (long)gridDim.x * blockDim.x) {
    long r  = idx;
    int kr   = (int)(r & 15);     r >>= 4;
    int lane = (int)(r & 31);     r >>= 5;
    int kt   = (int)(r % ktiles); r /= ktiles;
    int nt   = (int)(r % ntiles); r /= ntiles;
    int mat  = (int)r;
    int khalf = lane >> 4, nlq = lane & 15;
    int n = nt * 16 + nlq;
    int k = kt * 32 + (kr >> 3) * 16 + khalf * 8 + (kr & 7);
    float v = (k < Kdim) ? src[mat * srcMatStride + (long)k * Ndim + n] : 0.0f;
    dst[idx] = f2bf(v);
  }
}

// ---------------- fused GRU scan + dense VAE head ----------------
__global__ __launch_bounds__(256) void gru_vae_fused_kernel(
    const float* __restrict__ x,   const float* __restrict__ gru_b,
    const float* __restrict__ bd,  const float* __restrict__ bm,
    const float* __restrict__ bv,  const float* __restrict__ eps,
    const unsigned short* __restrict__ packed, float* __restrict__ out)
{
  __shared__ unsigned short h_s[2][16 * H_];  // double-buffered hidden state (bf16)
  __shared__ unsigned short hr_s[16 * H_];    // relu(h @ Wd + bd)        (bf16)
  __shared__ unsigned short x_s[16 * F_];     // current x_t tile          (bf16)
  __shared__ float zm_s[256];
  __shared__ float zl_s[256];

  const int tid  = threadIdx.x;
  const int lane = tid & 31;
  const int wv   = tid >> 5;       // 8 wave32s per workgroup
  const int nl   = lane & 15;      // A row / B,C column within 16-tile
  const int hlf  = lane >> 4;      // khalf for A, mhalf for C/D
  const int r0   = blockIdx.x * 16;

  // per-lane base pointers: every fragment load below is base + small constant,
  // so it folds into the 24-bit immediate offset of global_load_b128 / ds_load.
  const unsigned short* pW_l  = packed + OFF_W  + lane * 16;
  const unsigned short* pU_l  = packed + OFF_U  + lane * 16;
  const unsigned short* pWd_l = packed + OFF_WD + lane * 16;
  const unsigned short* pWm_l = packed + OFF_WM + lane * 16;
  const unsigned short* pWv_l = packed + OFF_WV + lane * 16;

  const unsigned short* xA  = &x_s[nl * F_ + hlf * 8];
  const unsigned short* hA0 = &h_s[0][nl * H_ + hlf * 8];
  const unsigned short* hA1 = &h_s[1][nl * H_ + hlf * 8];
  const unsigned short* rA  = &hr_s[nl * H_ + hlf * 8];

  for (int i = tid; i < 16 * H_; i += 256) { h_s[0][i] = 0; h_s[1][i] = 0; }
  __syncthreads();

  for (int s = 0; s < S_; ++s) {
    const int cur = s & 1;
    const unsigned short* hAc = cur ? hA1 : hA0;
    const unsigned short* hAn = cur ? hA0 : hA1;
    unsigned short* hWn = cur ? &h_s[0][0] : &h_s[1][0];
    const unsigned short* hRc = cur ? &h_s[1][0] : &h_s[0][0];

    // stage x_t tile (16 x 64) into LDS as bf16; prefetch next step's row
    {
      int m = tid >> 4, f0 = (tid & 15) * 4;
      const float* xp = x + ((long)(r0 + m) * S_ + s) * F_ + f0;
      float4 xv = *(const float4*)xp;
      unsigned short* d = &x_s[m * F_ + f0];
      d[0] = f2bf(xv.x); d[1] = f2bf(xv.y); d[2] = f2bf(xv.z); d[3] = f2bf(xv.w);
      if (s + 1 < S_) __builtin_prefetch(xp + F_, 0, 1);
    }
    __syncthreads();

    // ---- gate GEMMs: share each A fragment across the 3 gate accumulators
    for (int t = 0; t < 4; ++t) {
      const int j = wv + 8 * t;                  // hidden-column tile 0..31
      v8f ax[3] = {v8f{}, v8f{}, v8f{}};
      v8f ah[3] = {v8f{}, v8f{}, v8f{}};
      const unsigned short* pWj = pW_l + (long)j * 1024;   // + g*32768 + kt*512
      const unsigned short* pUj = pU_l + (long)j * 8192;   // + g*262144 + kt*512
#pragma unroll
      for (int kt = 0; kt < 2; ++kt) {
        Frag fa;
        loadA_lds(fa, xA + kt * 32);
#pragma unroll
        for (int g = 0; g < 3; ++g) {
          Frag fb;
          loadB_g(fb, pWj + g * 32768 + kt * 512);
          ax[g] = wmma_bf16(fa, fb, ax[g]);
        }
      }
#pragma unroll
      for (int kt = 0; kt < 16; ++kt) {
        Frag fa;
        loadA_lds(fa, hAc + kt * 32);
#pragma unroll
        for (int g = 0; g < 3; ++g) {
          Frag fb;
          loadB_g(fb, pUj + g * 262144 + kt * 512);
          ah[g] = wmma_bf16(fa, fb, ah[g]);
        }
      }
      const int colh = j * 16 + nl;
      const float bz  = gru_b[colh]        + gru_b[G3H + colh];
      const float br  = gru_b[512 + colh]  + gru_b[G3H + 512 + colh];
      const float bxh = gru_b[1024 + colh];
      const float bhh = gru_b[G3H + 1024 + colh];
#pragma unroll
      for (int i = 0; i < 8; ++i) {
        const int m = i + hlf * 8;
        float zg = sigmoidf_(ax[0][i] + ah[0][i] + bz);
        float rg = sigmoidf_(ax[1][i] + ah[1][i] + br);
        float cd = tanhf(ax[2][i] + bxh + rg * (ah[2][i] + bhh));
        float ho = bf2f(hRc[m * H_ + colh]);
        float hn = zg * ho + (1.0f - zg) * cd;
        hWn[m * H_ + colh] = f2bf(hn);
      }
    }
    __syncthreads();

    // ---- dense head: hr = relu(h_new @ Wd + bd)   (K = N = 512)
    for (int t = 0; t < 4; ++t) {
      const int ntile = wv + 8 * t;              // 0..31
      v8f a = {};
      const unsigned short* pWdj = pWd_l + (long)ntile * 8192;  // + kt*512
#pragma unroll
      for (int kt = 0; kt < 16; ++kt) {
        Frag fa, fb;
        loadA_lds(fa, hAn + kt * 32);
        loadB_g(fb, pWdj + kt * 512);
        a = wmma_bf16(fa, fb, a);
      }
      const int col = ntile * 16 + nl;
      const float b = bd[col];
#pragma unroll
      for (int i = 0; i < 8; ++i) {
        float v = a[i] + b;
        v = v > 0.0f ? v : 0.0f;
        hr_s[(i + hlf * 8) * H_ + col] = f2bf(v);
      }
    }
    __syncthreads();

    // ---- z_mean / z_logvar heads (K=512, N=16): wave0 -> Wm, wave1 -> Wv
    if (wv < 2) {
      const unsigned short* pB = (wv == 0) ? pWm_l : pWv_l;
      v8f a = {};
#pragma unroll
      for (int kt = 0; kt < 16; ++kt) {
        Frag fa, fb;
        loadA_lds(fa, rA + kt * 32);
        loadB_g(fb, pB + kt * 512);
        a = wmma_bf16(fa, fb, a);
      }
      const float b = (wv == 0) ? bm[nl] : bv[nl];
      float* dst = (wv == 0) ? zm_s : zl_s;
#pragma unroll
      for (int i = 0; i < 8; ++i)
        dst[(i + hlf * 8) * 16 + nl] = a[i] + b;
    }
    __syncthreads();

    // ---- reparameterize + write all three outputs for this (batch-tile, step)
    {
      int m = tid >> 4, n = tid & 15;
      long gi = ((long)(r0 + m) * S_ + s) * L_ + n;
      float zm = zm_s[m * 16 + n];
      float zl = zl_s[m * 16 + n];
      float e  = eps[gi];
      out[gi]            = zm;
      out[NTOT + gi]     = zl;
      out[2 * NTOT + gi] = zm + sqrtf(__expf(zl)) * e + 1e-4f;
    }
    __syncthreads();
  }
}

// ---------------- 20-step planar flow on z: one 16x16 tile per wave ----------
__global__ __launch_bounds__(256) void planar_flow_kernel(
    float* __restrict__ z, const float* __restrict__ b1,
    const unsigned short* __restrict__ pF1, const unsigned short* __restrict__ pF2)
{
  __shared__ float          zt[8][256];
  __shared__ unsigned short t1[8][256];
  const int lane = threadIdx.x & 31;
  const int wv   = threadIdx.x >> 5;
  const int nl   = lane & 15;
  const int hlf  = lane >> 4;
  float* zg = z + (((long)blockIdx.x * 8 + wv) * 16) * L_;

  const unsigned short* pF1_l = pF1 + lane * 16;   // + k*512
  const unsigned short* pF2_l = pF2 + lane * 16;

  {
    float4 a = *(const float4*)(zg + nl * 16 + hlf * 8);
    float4 b = *(const float4*)(zg + nl * 16 + hlf * 8 + 4);
    float* d = &zt[wv][nl * 16 + hlf * 8];
    d[0]=a.x; d[1]=a.y; d[2]=a.z; d[3]=a.w; d[4]=b.x; d[5]=b.y; d[6]=b.z; d[7]=b.w;
  }
  __syncthreads();

  for (int k = 0; k < KF_; ++k) {
    v8f zero = {};
    Frag fa, fb, fc, fd;
    {   // A = z tile, K padded 16 -> 32 with zeros (matches zero-padded B)
      const float* src = &zt[wv][nl * 16 + hlf * 8];
#pragma unroll
      for (int i = 0; i < 8; ++i)  fa.us[i] = f2bf(src[i]);
#pragma unroll
      for (int i = 8; i < 16; ++i) fa.us[i] = 0;
    }
    loadB_g(fb, pF1_l + k * 512);
    v8f a1 = wmma_bf16(fa, fb, zero);
    const float bb = b1[k * 16 + nl];
#pragma unroll
    for (int i = 0; i < 8; ++i)
      t1[wv][(i + hlf * 8) * 16 + nl] = f2bf(tanhf(a1[i] + bb));
    __syncthreads();

    fc.q[0] = *(const uint4*)&t1[wv][nl * 16 + hlf * 8];
    fc.q[1] = make_uint4(0u, 0u, 0u, 0u);
    loadB_g(fd, pF2_l + k * 512);
    v8f a2 = wmma_bf16(fc, fd, zero);
#pragma unroll
    for (int i = 0; i < 8; ++i)
      zt[wv][(i + hlf * 8) * 16 + nl] += a2[i];
    __syncthreads();
  }

  {
    const float* d = &zt[wv][nl * 16 + hlf * 8];
    *(float4*)(zg + nl * 16 + hlf * 8)     = make_float4(d[0], d[1], d[2], d[3]);
    *(float4*)(zg + nl * 16 + hlf * 8 + 4) = make_float4(d[4], d[5], d[6], d[7]);
  }
}

extern "C" void kernel_launch(void* const* d_in, const int* in_sizes, int n_in,
                              void* d_out, int out_size, void* d_ws, size_t ws_size,
                              hipStream_t stream) {
  (void)in_sizes; (void)n_in; (void)out_size; (void)ws_size;
  const float* x   = (const float*)d_in[0];
  const float* gW  = (const float*)d_in[1];
  const float* gU  = (const float*)d_in[2];
  const float* gb  = (const float*)d_in[3];
  const float* Wd  = (const float*)d_in[4];
  const float* bd  = (const float*)d_in[5];
  const float* Wm  = (const float*)d_in[6];
  const float* bm  = (const float*)d_in[7];
  const float* Wv  = (const float*)d_in[8];
  const float* bv  = (const float*)d_in[9];
  const float* eps = (const float*)d_in[10];
  const float* fW1 = (const float*)d_in[11];
  const float* fb1 = (const float*)d_in[12];
  const float* fW2 = (const float*)d_in[13];
  float* out = (float*)d_out;
  unsigned short* packed = (unsigned short*)d_ws;

  auto nb = [](long total) { return (int)((total + 255) / 256); };
  pack_frag_kernel<<<nb(96L * 2 * 512),  256, 0, stream>>>(gW,  packed + OFF_W,   64, 1536,  2, 96,  1, 0);
  pack_frag_kernel<<<nb(96L * 16 * 512), 256, 0, stream>>>(gU,  packed + OFF_U,  512, 1536, 16, 96,  1, 0);
  pack_frag_kernel<<<nb(32L * 16 * 512), 256, 0, stream>>>(Wd,  packed + OFF_WD, 512,  512, 16, 32,  1, 0);
  pack_frag_kernel<<<nb(1L  * 16 * 512), 256, 0, stream>>>(Wm,  packed + OFF_WM, 512,   16, 16,  1,  1, 0);
  pack_frag_kernel<<<nb(1L  * 16 * 512), 256, 0, stream>>>(Wv,  packed + OFF_WV, 512,   16, 16,  1,  1, 0);
  pack_frag_kernel<<<nb(20L * 1 * 512),  256, 0, stream>>>(fW1, packed + OFF_F1,  16,   16,  1,  1, 20, 256);
  pack_frag_kernel<<<nb(20L * 1 * 512),  256, 0, stream>>>(fW2, packed + OFF_F2,  16,   16,  1,  1, 20, 256);

  gru_vae_fused_kernel<<<B_ / 16, 256, 0, stream>>>(x, gb, bd, bm, bv, eps, packed, out);
  planar_flow_kernel<<<(B_ * S_ / 16) / 8, 256, 0, stream>>>(out + 2L * NTOT, fb1,
                                                             packed + OFF_F1, packed + OFF_F2);
}
